// GcnNet_558345748855
// MI455X (gfx1250) — matrix operand order repaired and checked by reading, not compiled, not run on hardware
//
#include <hip/hip_runtime.h>
#include <hip/hip_bf16.h>

typedef __attribute__((ext_vector_type(16))) _Float16 v16h;
typedef __attribute__((ext_vector_type(8)))  _Float16 v8h;
typedef __attribute__((ext_vector_type(8)))  float    v8f;

#define N_NODES 16384
#define NFEAT   128
#define NHID1   128
#define NHID2   64

// ---------------------------------------------------------------------------
// K1: S1t[j][i] = sum_k X[i][k] * W1[k][j]   (f16, transposed layout)
// 1024 blocks x 128 threads; each block handles 16 rows of X.
// ---------------------------------------------------------------------------
__global__ __launch_bounds__(128) void xw1_kernel(const float* __restrict__ X,
                                                  const float* __restrict__ W1,
                                                  _Float16* __restrict__ S1t) {
  __shared__ float xs[16][NFEAT];
  const int b = blockIdx.x;
  const int t = threadIdx.x;
  for (int r = 0; r < 16; ++r)
    xs[r][t] = X[((size_t)b * 16 + r) * NFEAT + t];
  __syncthreads();
  float acc[16];
  #pragma unroll
  for (int r = 0; r < 16; ++r) acc[r] = 0.f;
  for (int k = 0; k < NFEAT; ++k) {
    float w = W1[k * NHID1 + t];
    #pragma unroll
    for (int r = 0; r < 16; ++r) acc[r] = fmaf(xs[r][k], w, acc[r]);
  }
  for (int r = 0; r < 16; ++r)
    S1t[(size_t)t * N_NODES + b * 16 + r] = (_Float16)acc[r];
}

// ---------------------------------------------------------------------------
// K3: S2t[j][i] = sum_k h[i][k] * W2[k][j]   (f16, transposed layout)
// 1024 blocks x 64 threads; each block handles 16 rows of h.
// ---------------------------------------------------------------------------
__global__ __launch_bounds__(64) void hw2_kernel(const _Float16* __restrict__ h,
                                                 const float* __restrict__ W2,
                                                 _Float16* __restrict__ S2t) {
  __shared__ _Float16 hs[16][NHID1];
  const int b = blockIdx.x;
  const int t = threadIdx.x;
  for (int i = t; i < 16 * NHID1; i += 64)
    (&hs[0][0])[i] = h[(size_t)b * 16 * NHID1 + i];
  __syncthreads();
  float acc[16];
  #pragma unroll
  for (int r = 0; r < 16; ++r) acc[r] = 0.f;
  for (int k = 0; k < NHID1; ++k) {
    float w = W2[k * NHID2 + t];
    #pragma unroll
    for (int r = 0; r < 16; ++r) acc[r] = fmaf((float)hs[r][k], w, acc[r]);
  }
  for (int r = 0; r < 16; ++r)
    S2t[(size_t)t * N_NODES + b * 16 + r] = (_Float16)acc[r];
}

// ---------------------------------------------------------------------------
// K2/K4: partial[y][m][n] = sum_{k in half y} adj[m][k] * Bt[n][k]
// WMMA f16 -> f32. BM=256, BK=32, BN=NHID. 256 threads = 8 waves: 4(M) x 2(N).
// adj tiles converted fp32->f16 through LDS; Bt staged column-major in LDS.
// ---------------------------------------------------------------------------
template <int NHID>
__global__ __launch_bounds__(256) void adj_gemm_kernel(const float* __restrict__ adj,
                                                       const _Float16* __restrict__ Bt,
                                                       float* __restrict__ partial) {
  constexpr int NTW = NHID / 32;   // n-tiles per wave (N split across 2 waves)
  constexpr int LDA = 40;          // padded f16 stride (80B, multiple of 16B)
  __shared__ __align__(16) _Float16 Atile[256 * LDA];
  __shared__ __align__(16) _Float16 Btile[NHID * LDA];

  const int t    = threadIdx.x;
  const int wave = t >> 5;
  const int lane = t & 31;
  const int l15  = lane & 15;
  const int wm   = wave >> 1;          // 0..3 -> 64 rows each
  const int wn   = wave & 1;           // 0..1 -> NHID/2 cols each
  const int m0   = blockIdx.x * 256;
  const int kbase = blockIdx.y * (N_NODES / 2);
  const int khalf = (lane < 16) ? 0 : 16;   // B fragment K-half
  const int kb    = (lane < 16) ? 0 : 16;   // A fragment K-subblock

  v8f acc[4][NTW];
  const v8f vzero = {0.f, 0.f, 0.f, 0.f, 0.f, 0.f, 0.f, 0.f};
  #pragma unroll
  for (int mt = 0; mt < 4; ++mt)
    #pragma unroll
    for (int nt = 0; nt < NTW; ++nt) acc[mt][nt] = vzero;

  for (int k0 = 0; k0 < N_NODES / 2; k0 += 32) {
    // ---- stage A: 256 rows x 32 cols of adj, fp32 -> f16 into LDS
    {
      const float4* src = (const float4*)(adj + (size_t)(m0 + t) * N_NODES + kbase + k0);
      _Float16* dst = &Atile[t * LDA];
      #pragma unroll
      for (int c = 0; c < 2; ++c) {
        float4 f0 = src[4 * c + 0], f1 = src[4 * c + 1];
        float4 f2 = src[4 * c + 2], f3 = src[4 * c + 3];
        v8h h0 = {(_Float16)f0.x, (_Float16)f0.y, (_Float16)f0.z, (_Float16)f0.w,
                  (_Float16)f1.x, (_Float16)f1.y, (_Float16)f1.z, (_Float16)f1.w};
        v8h h1 = {(_Float16)f2.x, (_Float16)f2.y, (_Float16)f2.z, (_Float16)f2.w,
                  (_Float16)f3.x, (_Float16)f3.y, (_Float16)f3.z, (_Float16)f3.w};
        *(v8h*)(dst + 16 * c)     = h0;
        *(v8h*)(dst + 16 * c + 8) = h1;
      }
      __builtin_prefetch(src + 8, 0, 0);   // next k-tile of this row
    }
    // ---- stage B: Btile[col][k] = Bt[col][kbase+k0+k], column-major in LDS
    if (t < NHID * 2) {
      const int col = t >> 1;
      const int ko  = (t & 1) * 16;
      const v8h* src = (const v8h*)(Bt + (size_t)col * N_NODES + kbase + k0 + ko);
      v8h b0 = src[0], b1 = src[1];
      *(v8h*)&Btile[col * LDA + ko]     = b0;
      *(v8h*)&Btile[col * LDA + ko + 8] = b1;
    }
    __syncthreads();

    // ---- B fragments (K contiguous per lane, k = khalf + s, same order as A)
    v16h bfrag[NTW];
    #pragma unroll
    for (int nt = 0; nt < NTW; ++nt) {
      const int col = wn * (NTW * 16) + nt * 16 + l15;
      const v8h* p = (const v8h*)&Btile[col * LDA + khalf];
      bfrag[nt] = __builtin_shufflevector(p[0], p[1],
          0, 1, 2, 3, 4, 5, 6, 7, 8, 9, 10, 11, 12, 13, 14, 15);
    }
    // ---- A fragments + WMMA
    #pragma unroll
    for (int mt = 0; mt < 4; ++mt) {
      const int row = wm * 64 + mt * 16 + l15;
      const v8h* p = (const v8h*)&Atile[row * LDA + kb];   // K = kb..kb+15
      v16h afrag = __builtin_shufflevector(p[0], p[1],
          0, 1, 2, 3, 4, 5, 6, 7, 8, 9, 10, 11, 12, 13, 14, 15);
      #pragma unroll
      for (int nt = 0; nt < NTW; ++nt)
        acc[mt][nt] = __builtin_amdgcn_wmma_f32_16x16x32_f16(
            false, afrag, false, bfrag[nt], (short)0, acc[mt][nt], false, false);
    }
    __syncthreads();
  }

  // ---- write f32 partial sums
  float* pout = partial + (size_t)blockIdx.y * N_NODES * NHID;
  #pragma unroll
  for (int mt = 0; mt < 4; ++mt)
    #pragma unroll
    for (int nt = 0; nt < NTW; ++nt) {
      const int col = wn * (NTW * 16) + nt * 16 + l15;
      #pragma unroll
      for (int r = 0; r < 8; ++r) {
        const int row = m0 + wm * 64 + mt * 16 + ((lane < 16) ? r : r + 8);
        pout[(size_t)row * NHID + col] = acc[mt][nt][r];
      }
    }
}

// ---------------------------------------------------------------------------
// Combine K-split partials + bias (+ReLU); f16 out (layer1) or f32 out (layer2)
// ---------------------------------------------------------------------------
template <bool RELU, bool F16OUT>
__global__ __launch_bounds__(256) void combine_kernel(const float* __restrict__ p,
                                                      const float* __restrict__ bias,
                                                      int nhid,
                                                      _Float16* __restrict__ hout,
                                                      float* __restrict__ fout) {
  const size_t i = (size_t)blockIdx.x * 256 + threadIdx.x;
  const int col = (int)(i & (size_t)(nhid - 1));
  float v = p[i] + p[(size_t)N_NODES * nhid + i] + bias[col];
  if (RELU) v = fmaxf(v, 0.f);
  if (F16OUT) hout[i] = (_Float16)v;
  else        fout[i] = v;
}

// ---------------------------------------------------------------------------
extern "C" void kernel_launch(void* const* d_in, const int* in_sizes, int n_in,
                              void* d_out, int out_size, void* d_ws, size_t ws_size,
                              hipStream_t stream) {
  const float* feature = (const float*)d_in[0];
  const float* adj     = (const float*)d_in[1];
  const float* W1      = (const float*)d_in[2];
  const float* b1      = (const float*)d_in[3];
  const float* W2      = (const float*)d_in[4];
  const float* b2      = (const float*)d_in[5];
  float* out = (float*)d_out;

  char* ws = (char*)d_ws;
  _Float16* S1t  = (_Float16*)(ws);                       // 128*16384*2  = 4 MiB
  _Float16* S2t  = (_Float16*)(ws + ((size_t)4 << 20));   // 64*16384*2   = 2 MiB
  _Float16* hbuf = (_Float16*)(ws + ((size_t)6 << 20));   // 16384*128*2  = 4 MiB
  float*  partial = (float*)  (ws + ((size_t)16 << 20));  // 2*16384*128*4 = 16 MiB

  // layer 1
  xw1_kernel<<<N_NODES / 16, 128, 0, stream>>>(feature, W1, S1t);
  adj_gemm_kernel<NHID1><<<dim3(N_NODES / 256, 2), 256, 0, stream>>>(adj, S1t, partial);
  combine_kernel<true, true><<<(N_NODES * NHID1) / 256, 256, 0, stream>>>(
      partial, b1, NHID1, hbuf, nullptr);

  // layer 2
  hw2_kernel<<<N_NODES / 16, 64, 0, stream>>>(hbuf, W2, S2t);
  adj_gemm_kernel<NHID2><<<dim3(N_NODES / 256, 2), 256, 0, stream>>>(adj, S2t, partial);
  combine_kernel<false, false><<<(N_NODES * NHID2) / 256, 256, 0, stream>>>(
      partial, b2, NHID2, nullptr, out);
}
